// DistTokenMix_9758165696675
// MI455X (gfx1250) — matrix operand order repaired and checked by reading, not compiled, not run on hardware
//
#include <hip/hip_runtime.h>

// DistTokenMix on MI455X (gfx1250):
// out[b,i,d] = sum_j alpha[spd[i,j], d] * h[b,j,d]
//            = sum_{dist} ( M_dist @ h[b] ) * alpha[dist]   (M_dist = (spd==dist))
//
// Per-bucket f32 WMMA accumulators; masks are exact bf16 0/1 built in 2 VALU
// per A-VGPR per bucket (compiler fuses (lo|hi)&sel into one v_bitop3, then
// *(0x3F80>>d)). Since the masks partition the plane (sum_d M_d = ones),
// bucket 7 is recovered from column sums: S_7 = colsum(h) - sum_{d<7} S_d,
// folded into the alpha combine as out = sum_{d<7} S_d*(a_d-a_7) + colsum*a_7.
// A deterministic prepass computes colsum over bf16-truncated h (matching the
// main kernel's v_perm truncation) into ws. Mask fragments are TRIPLE-buffered
// so the bucket d+2 mask writes clear the bf16-WMMA WAR hazard window (4
// co-exec slots) without v_nops.

typedef __attribute__((ext_vector_type(16))) __bf16       v16bf;
typedef __attribute__((ext_vector_type(8)))  float        v8f;
typedef __attribute__((ext_vector_type(8)))  unsigned int v8u;

namespace {
constexpr int Bn  = 8;     // batches
constexpr int Nn  = 4096;  // tokens
constexpr int Dn  = 64;    // channels
constexpr int NQ  = 4;     // j-quarters for colsum partials
constexpr size_t kWsNeed = (size_t)NQ * Bn * Dn * sizeof(float);  // 8 KB
}

// ---- Prepass: ws[q][b][c] = sum_{j in quarter q} bf16_trunc(h[b,j,c]) ------
__global__ __launch_bounds__(512, 1) void colsum_partials(
    const float* __restrict__ h, float* __restrict__ ws)
{
  const int b   = blockIdx.x;        // 0..7
  const int q   = blockIdx.y;        // 0..3
  const int c   = threadIdx.x & 63;  // channel
  const int sub = threadIdx.x >> 6;  // 0..7 sub-range of j

  __shared__ float red[8][Dn];

  const float* hp = h + ((long)b * Nn + q * (Nn / NQ) + sub * 128) * Dn + c;
  float s = 0.0f;
  #pragma unroll 4
  for (int j = 0; j < 128; ++j) {
    const unsigned int u = __float_as_uint(hp[(long)j * Dn]) & 0xFFFF0000u;
    s += __uint_as_float(u);   // same truncation as the main kernel's v_perm
  }
  red[sub][c] = s;
  __syncthreads();
  if (sub == 0) {
    float tot = 0.0f;
    #pragma unroll
    for (int k = 0; k < 8; ++k) tot += red[k][c];   // fixed order: determinism
    ws[(q * Bn + b) * Dn + c] = tot;
  }
}

// ---- Main kernel -----------------------------------------------------------
template <bool USE_COLSUM>
__global__ __launch_bounds__(512, 1) void dist_token_mix_wmma(
    const float* __restrict__ h,      // [B, N, D] f32
    const int*   __restrict__ spd,    // [N, N] int32, values in [0, 8)
    const float* __restrict__ alpha,  // [8, D] f32
    const float* __restrict__ csum,   // ws partials [NQ, B, D] (if USE_COLSUM)
    float*       __restrict__ out)    // [B, N, D] f32
{
  constexpr int ND = USE_COLSUM ? 7 : 8;   // buckets materialized as masks

  const int lane = threadIdx.x & 31;
  const int w    = threadIdx.x >> 5;   // 16 waves
  const int b    = w & 7;              // batch owned by this wave
  const int ntp  = w >> 3;             // which 32-channel half
  const int c0   = ntp * 32;           // channel base; tiles at c0 and c0+16
  const int i0   = blockIdx.x * 16;    // output row tile
  const int m    = lane & 15;
  const int hi   = lane >> 4;

  v8f acc[ND][2];
  #pragma unroll
  for (int d = 0; d < ND; ++d)
    #pragma unroll
    for (int t = 0; t < 2; ++t)
      #pragma unroll
      for (int v = 0; v < 8; ++v) acc[d][t][v] = 0.0f;

  // A-fragment (16x32 bf16): lane m holds row i0+m; lanes 0-15: K 0..7,16..23;
  // lanes 16-31: K 8..15, 24..31 (per ISA table).
  const long spdRow = (long)(i0 + m) * (long)Nn;
  const int  kbA    = hi * 8;
  // B-fragment (32x16 bf16): lane n holds column; lanes 0-15: K 0..7,16..23;
  // lanes 16-31: K 8..15, 24..31 (same order as A).
  const int  kbB    = hi * 8;
  const long hB     = (long)b * Nn * Dn;

  #pragma unroll 1
  for (int jt = 0; jt < Nn / 32; ++jt) {
    const int j0 = jt * 32;

    const int* sp = spd + spdRow + j0 + kbA;
    const int4 ra0 = ((const int4*)sp)[0];
    const int4 ra1 = ((const int4*)sp)[1];
    const int4 rb0 = ((const int4*)(sp + 16))[0];
    const int4 rb1 = ((const int4*)(sp + 16))[1];

    __builtin_prefetch(sp + 32, 0, 1);  // next spd tile (speculative)

    // One-hot bit-pack: s[v] bit d = (spd(k0)==d), bit 16+d = (spd(k1)==d).
    // (The compiler keeps the halves separate and fuses the OR into the
    //  per-bucket select as v_bitop3.)
    unsigned int s[8];
    s[0] = (0x10000u << ra0.y) | (1u << ra0.x);
    s[1] = (0x10000u << ra0.w) | (1u << ra0.z);
    s[2] = (0x10000u << ra1.y) | (1u << ra1.x);
    s[3] = (0x10000u << ra1.w) | (1u << ra1.z);
    s[4] = (0x10000u << rb0.y) | (1u << rb0.x);
    s[5] = (0x10000u << rb0.w) | (1u << rb0.z);
    s[6] = (0x10000u << rb1.y) | (1u << rb1.x);
    s[7] = (0x10000u << rb1.w) | (1u << rb1.z);

    // B fragments: h in bf16 via high-half perm (truncation).
    v8u bu[2];
    #pragma unroll
    for (int t = 0; t < 2; ++t) {
      const float* hp = h + hB + (long)(j0 + kbB) * Dn + (c0 + t * 16 + m);
      #pragma unroll
      for (int v = 0; v < 8; ++v) {
        const unsigned int lo = __float_as_uint(hp[(2 * v + (v >= 4 ? 8 : 0)) * Dn]);
        const unsigned int hh = __float_as_uint(hp[(2 * v + 1 + (v >= 4 ? 8 : 0)) * Dn]);
        bu[t][v] = __builtin_amdgcn_perm(hh, lo, 0x07060302u);
      }
    }
    const v16bf B0 = __builtin_bit_cast(v16bf, bu[0]);
    const v16bf B1 = __builtin_bit_cast(v16bf, bu[1]);

    // Buckets: 2-op mask build; triple-buffered A fragments so the buffer
    // reuse distance (~16 VALU + 4 WMMA) exceeds the WAR hazard window.
    v8u mu[3];
    #pragma unroll
    for (int d = 0; d < ND; ++d) {
      const unsigned int selBits = 0x00010001u << d;  // bits d and 16+d
      const unsigned int oneBf   = 0x3F80u >> d;      // (1<<d)*oneBf == 0x3F80
      v8u& mw = mu[d % 3];
      #pragma unroll
      for (int v = 0; v < 8; ++v)
        mw[v] = (s[v] & selBits) * oneBf;             // bf16 1.0 per hit
      const v16bf A = __builtin_bit_cast(v16bf, mw);
      acc[d][0] = __builtin_amdgcn_wmma_f32_16x16x32_bf16(
          false, A, false, B0, (short)0, acc[d][0], false, false);
      acc[d][1] = __builtin_amdgcn_wmma_f32_16x16x32_bf16(
          false, A, false, B1, (short)0, acc[d][1], false, false);
    }
  }

  // ---- epilogue: combine buckets with alpha (C/D layout: lane = column) ----
  float o[2][8];
  #pragma unroll
  for (int t = 0; t < 2; ++t)
    #pragma unroll
    for (int v = 0; v < 8; ++v) o[t][v] = 0.0f;

  if constexpr (USE_COLSUM) {
    #pragma unroll
    for (int t = 0; t < 2; ++t) {
      const int   c  = c0 + t * 16 + m;
      const float a7 = alpha[7 * Dn + c];
      float cs = 0.0f;
      #pragma unroll
      for (int q = 0; q < NQ; ++q) cs += csum[(q * Bn + b) * Dn + c];
      #pragma unroll
      for (int v = 0; v < 8; ++v) o[t][v] = cs * a7;  // colsum * alpha_7
      #pragma unroll
      for (int d = 0; d < 7; ++d) {
        const float ad = alpha[d * Dn + c] - a7;      // (alpha_d - alpha_7)
        #pragma unroll
        for (int v = 0; v < 8; ++v) o[t][v] += acc[d][t][v] * ad;
      }
    }
  } else {
    #pragma unroll
    for (int d = 0; d < 8; ++d) {
      const float a0 = alpha[d * Dn + c0 + m];
      const float a1 = alpha[d * Dn + c0 + 16 + m];
      #pragma unroll
      for (int v = 0; v < 8; ++v) {
        o[0][v] += acc[d][0][v] * a0;
        o[1][v] += acc[d][1][v] * a1;
      }
    }
  }

  // ---- store: row = i0 + v + 8*hi, col = lane&15 (coalesced 64B runs) ------
  #pragma unroll
  for (int t = 0; t < 2; ++t) {
    const int c = c0 + t * 16 + m;
    #pragma unroll
    for (int v = 0; v < 8; ++v) {
      const int row = i0 + v + 8 * hi;
      out[((long)b * Nn + row) * Dn + c] = o[t][v];
    }
  }
}

extern "C" void kernel_launch(void* const* d_in, const int* in_sizes, int n_in,
                              void* d_out, int out_size, void* d_ws, size_t ws_size,
                              hipStream_t stream) {
  (void)in_sizes; (void)n_in; (void)out_size;
  const float* h     = (const float*)d_in[0];
  const int*   spd   = (const int*)d_in[1];
  const float* alpha = (const float*)d_in[2];
  float*       out   = (float*)d_out;

  dim3 grid(Nn / 16);   // 256 row tiles
  dim3 block(512);      // 16 wave32 waves: 8 batches x 2 channel-halves

  if (ws_size >= kWsNeed && d_ws != nullptr) {
    float* ws = (float*)d_ws;
    hipLaunchKernelGGL(colsum_partials, dim3(Bn, NQ), dim3(512), 0, stream,
                       h, ws);
    hipLaunchKernelGGL((dist_token_mix_wmma<true>), grid, block, 0, stream,
                       h, spd, alpha, (const float*)ws, out);
  } else {
    hipLaunchKernelGGL((dist_token_mix_wmma<false>), grid, block, 0, stream,
                       h, spd, alpha, (const float*)nullptr, out);
  }
}
